// TransformerDecoder_59038620451485
// MI455X (gfx1250) — matrix operand order repaired and checked by reading, not compiled, not run on hardware
//
#include <hip/hip_runtime.h>

// ---------------------------------------------------------------------------
// CDNA5 / gfx1250 implementation of the transformer-decoder block.
//   * all matmuls on v_wmma_f32_16x16x32_bf16 (fp32 accumulate)
//   * flash-style attention (online softmax) -> ~18MB workspace, L2-resident
//   * K tiles staged LDS<-global by the Tensor Data Mover (tensor_load_to_lds,
//     TENSORcnt) ; W tiles staged by global_load_async_to_lds_b128 (ASYNCcnt)
// Roofline: ~23 GFLOP vs ~52MB traffic -> a few microseconds either way; the
// wins are bf16 WMMA everywhere, no materialized 1024x8192 score matrix, and
// DMA staging that keeps VALU free for cvt/softmax work.
// ---------------------------------------------------------------------------

typedef __bf16 bf16;
typedef __attribute__((ext_vector_type(16))) __bf16 v16bf;
typedef __attribute__((ext_vector_type(8)))  float  v8f;

__device__ __forceinline__ v8f zero8() {
  v8f z;
#pragma unroll
  for (int i = 0; i < 8; ++i) z[i] = 0.f;
  return z;
}

// A-matrix (16x32, bf16) per-lane k pattern (contiguous, matches B):
//   lane = m + 16h ; element j -> k = 16h + j
__device__ __forceinline__ int kpatA(int j, int h) {
  return 16 * h + j;
}

union BfVec { v16bf v; uint4 q[2]; };

// contiguous 16 x bf16 (32B) load from LDS, 16B-aligned pieces
__device__ __forceinline__ v16bf lds_load16(const bf16* p) {
  BfVec u;
  u.q[0] = ((const uint4*)p)[0];
  u.q[1] = ((const uint4*)p)[1];
  return u.v;
}

__device__ __forceinline__ v8f wmma_bf16(v16bf a, v16bf b, v8f c) {
  return __builtin_amdgcn_wmma_f32_16x16x32_bf16(
      /*neg_a=*/false, a, /*neg_b=*/false, b,
      /*c_mod=*/(short)0, c, /*reuse_a=*/false, /*reuse_b=*/false);
}

// LDS byte offset of a __shared__ object: the LDS aperture maps flat
// addr[31:0] directly onto the wave's LDS allocation (ISA 10.2).
__device__ __forceinline__ unsigned lds_off_of(const void* p) {
  return (unsigned)(size_t)p;
}

// Async DMA: 16B per lane, global -> LDS, tracked with ASYNCcnt (ISA 10.7).
__device__ __forceinline__ void async_load_b128(unsigned lds_off,
                                                const void* gptr) {
  asm volatile("global_load_async_to_lds_b128 %0, %1, off"
               :: "v"(lds_off), "v"((unsigned long long)(size_t)gptr)
               : "memory");
}
__device__ __forceinline__ void wait_async0() {
  asm volatile("s_wait_asynccnt 0x0" ::: "memory");
}

// Tensor Data Mover: DMA `bytes16k` = tile_dim0 contiguous 2B elements from
// `gaddr` into LDS at `lds_off`.  D# built per ISA 8.3/8.4:
//   group0: count=1 | lds_addr | global_addr | type=2
//   group1: data_size=1(2B), tensor_dim0=tile_dim0=nelem, tensor_dim1=1,
//           tile_dim1=1, dim0_stride=nelem  (pure linear copy)
__device__ __forceinline__ void tdm_load_linear_b16(unsigned lds_off,
                                                    unsigned long long gaddr,
                                                    unsigned nelem) {
  unsigned g0w0 = 1u;                                   // count=1
  unsigned g0w1 = lds_off;                              // lds_addr
  unsigned g0w2 = (unsigned)gaddr;                      // global_addr[31:0]
  unsigned g0w3 = (unsigned)((gaddr >> 32) & 0x01FFFFFFull) | (2u << 30);
  unsigned g1w0 = 0x00010000u;                          // data_size=1 (2B)
  unsigned g1w1 = (nelem & 0xFFFFu) << 16;              // tensor_dim0 lo
  unsigned g1w2 = (nelem >> 16) | (1u << 16);           // dim0 hi | dim1=1
  unsigned g1w3 = (nelem & 0xFFFFu) << 16;              // tile_dim0
  unsigned g1w4 = 1u;                                   // tile_dim1=1
  unsigned g1w5 = nelem;                                // dim0_stride
  unsigned g1w6 = 0u, g1w7 = 0u;
  asm volatile(
      "s_mov_b32 s60, %0\n\t"
      "s_mov_b32 s61, %1\n\t"
      "s_mov_b32 s62, %2\n\t"
      "s_mov_b32 s63, %3\n\t"
      "s_mov_b32 s64, %4\n\t"
      "s_mov_b32 s65, %5\n\t"
      "s_mov_b32 s66, %6\n\t"
      "s_mov_b32 s67, %7\n\t"
      "s_mov_b32 s68, %8\n\t"
      "s_mov_b32 s69, %9\n\t"
      "s_mov_b32 s70, %10\n\t"
      "s_mov_b32 s71, %11\n\t"
      "tensor_load_to_lds s[60:63], s[64:71]"
      :: "s"(g0w0), "s"(g0w1), "s"(g0w2), "s"(g0w3),
         "s"(g1w0), "s"(g1w1), "s"(g1w2), "s"(g1w3),
         "s"(g1w4), "s"(g1w5), "s"(g1w6), "s"(g1w7)
      : "s60","s61","s62","s63","s64","s65","s66","s67",
        "s68","s69","s70","s71","memory");
}

// ---------------------------------------------------------------------------
// proj_l2: out[s][row][0..127] = l2norm( X[s,row,:] @ W^T + bias ), bf16 out.
// X element (s,row,ch) lives at  src + t*tStride + s*sStride + ch*hw + p,
// with row = t*hw + p (channel-major source layouts: tgt / memory / Y).
// Block: 128 threads = 4 waves; wave -> 16 rows x 128 cols via 8x8 WMMAs.
// The 128x32 fp32 W slice is DMAed once per block into LDS (ASYNCcnt) and
// shared by all 4 waves instead of 4x redundant global reads.
// ---------------------------------------------------------------------------
__global__ void __launch_bounds__(128)
proj_l2_kernel(const float* __restrict__ src, const float* __restrict__ W,
               const float* __restrict__ bias, bf16* __restrict__ out,
               int hw, long tStride, long sStride, long outSB) {
  __shared__ __align__(16) float Wlds[128 * 32];   // 16KB fp32 W slice

  const int s    = blockIdx.y;
  const int tid  = threadIdx.x;
  const int lane = tid & 31;
  const int wv   = tid >> 5;
  const int ln   = lane & 15, lh = lane >> 4;
  const int r0   = blockIdx.x * 64 + wv * 16;
  const int row  = r0 + ln;                 // A-matrix row owned by this lane
  const int t    = row / hw, p = row % hw;
  const float* abase = src + (long)t * tStride + (long)s * sStride + p;

  v8f acc[8];
#pragma unroll
  for (int nt = 0; nt < 8; ++nt) acc[nt] = zero8();

  for (int ko = 0; ko < 256; ko += 32) {
    // --- async-DMA W[:, ko:ko+32] into LDS (1024 x 16B chunks) ---
    for (int e = tid; e < 1024; e += 128) {
      int rown = e >> 3, chn = (e & 7) * 4;
      async_load_b128(lds_off_of(Wlds + rown * 32 + chn),
                      W + (long)rown * 256 + ko + chn);
    }

    // --- A fragment: strided global gathers (overlaps with the DMA) ---
    v16bf a;
#pragma unroll
    for (int j = 0; j < 16; ++j)
      a[j] = (bf16)abase[(long)(ko + kpatA(j, lh)) * hw];

    wait_async0();
    __syncthreads();

#pragma unroll
    for (int nt = 0; nt < 8; ++nt) {
      // B[k][n] = W[nt*16+n][ko+k] ; lane n=ln reads k = 16h + j contiguous
      const float* wrow = Wlds + (nt * 16 + ln) * 32 + 16 * lh;
      v16bf b;
#pragma unroll
      for (int j = 0; j < 16; ++j) b[j] = (bf16)wrow[j];
      acc[nt] = wmma_bf16(a, b, acc[nt]);
    }
    __syncthreads();   // protect Wlds before next DMA
  }

  float bv[8];
#pragma unroll
  for (int nt = 0; nt < 8; ++nt) bv[nt] = bias[nt * 16 + ln];

  bf16* ob = out + (long)s * outSB;
#pragma unroll
  for (int r = 0; r < 8; ++r) {           // output row = r0 + r + 8*lh
    float q = 0.f;
#pragma unroll
    for (int nt = 0; nt < 8; ++nt) {
      float v = acc[nt][r] + bv[nt];
      acc[nt][r] = v;
      q += v * v;
    }
#pragma unroll
    for (int off = 1; off < 16; off <<= 1) q += __shfl_xor(q, off);
    float rn = 1.f / fmaxf(sqrtf(q), 1e-12f);
    const long orow = (long)(r0 + r + 8 * lh) * 128;
#pragma unroll
    for (int nt = 0; nt < 8; ++nt)
      ob[orow + nt * 16 + ln] = (bf16)(acc[nt][r] * rn);
  }
}

// ---------------------------------------------------------------------------
// Flash attention: O[s] = softmax(30 * Q K^T) @ V,  kd = 128, dv = DV.
// Q,K: row-major bf16 (L,128) in workspace. V: fp32 channel-major global
// (V[t][s][dv][p]) -> staged *transposed* to LDS with contiguous loads.
// K tiles (8KB) are DMAed by the Tensor Data Mover (wave 0 issues, TENSORcnt).
// Output fp32 channel-major: O + s*oSS + dv*hw + row.
// Block: 128 threads = 4 waves, wave = 16 query rows, key tiles of 32.
// ---------------------------------------------------------------------------
template <int DV>
__global__ void __launch_bounds__(128)
attn_kernel(const bf16* __restrict__ Q, const bf16* __restrict__ K,
            const float* __restrict__ Vsrc, float* __restrict__ O,
            int Lk, long qSS, long kSS, long vTS, long vSS, int hw, long oSS) {
  __shared__ __align__(16) bf16 Klds[32 * 128];   // K tile, row-major
  __shared__ __align__(16) bf16 Vlds[DV * 32];    // V tile, transposed [dv][k]
  __shared__ __align__(16) bf16 Plds[4 * 16 * 32];// per-wave P scratch

  const int s    = blockIdx.y;
  const int tid  = threadIdx.x;
  const int lane = tid & 31, wv = tid >> 5;
  const int ln   = lane & 15, lh = lane >> 4;
  const int q0   = blockIdx.x * 64 + wv * 16;
  const float SCALE = 30.0f;                      // 1/TAU
  const unsigned KldsOff = lds_off_of(Klds);

  // Preload Q fragments (16 rows x kd=128 -> 4 A fragments), reused all tiles
  const bf16* qp = Q + (long)s * qSS + (long)(q0 + ln) * 128;
  v16bf qf[4];
#pragma unroll
  for (int kk = 0; kk < 4; ++kk)
#pragma unroll
    for (int j = 0; j < 16; ++j) qf[kk][j] = qp[kk * 32 + kpatA(j, lh)];

  v8f o[DV / 16];
#pragma unroll
  for (int nt = 0; nt < DV / 16; ++nt) o[nt] = zero8();
  float mrun[8], lrun[8];
#pragma unroll
  for (int r = 0; r < 8; ++r) { mrun[r] = -1e30f; lrun[r] = 0.f; }

  const bf16* kbase = K + (long)s * kSS;
  bf16* pl = Plds + wv * (16 * 32);

  for (int k0 = 0; k0 < Lk; k0 += 32) {
    // --- TDM: wave0 DMAs the 32x128 bf16 K tile straight into LDS ---
    if (wv == 0)
      tdm_load_linear_b16(KldsOff,
                          (unsigned long long)(size_t)(kbase + (long)k0 * 128),
                          32u * 128u);

    // --- stage V tile transposed: Vlds[dv][k] = Vsrc[t,s,dv,p0+k] ---
    {
      int t = k0 / hw, p0 = k0 % hw;
      const float* vb = Vsrc + (long)t * vTS + (long)s * vSS + p0;
      for (int dv = tid; dv < DV; dv += 128) {
        const float* vp = vb + (long)dv * hw;   // 32 contiguous fp32
        bf16* dst = Vlds + dv * 32;
#pragma unroll
        for (int i = 0; i < 32; i += 4) {
          float4 f = *(const float4*)(vp + i);
          dst[i + 0] = (bf16)f.x; dst[i + 1] = (bf16)f.y;
          dst[i + 2] = (bf16)f.z; dst[i + 3] = (bf16)f.w;
        }
      }
      // prefetch next iteration's V rows into cache
      if (k0 + 32 < Lk) {
        int tn = (k0 + 32) / hw, pn = (k0 + 32) % hw;
        __builtin_prefetch(Vsrc + (long)tn * vTS + (long)s * vSS + pn +
                               (long)(tid & (DV - 1)) * hw, 0, 0);
      }
    }
    __builtin_amdgcn_s_wait_tensorcnt(0);   // drains wave0's TDM; no-op others
    __syncthreads();

    // --- scores S = Q K^T for 16 rows x 32 keys (two 16-col fragments) ---
    v8f s0 = zero8(), s1 = zero8();
#pragma unroll
    for (int kk = 0; kk < 4; ++kk) {
      v16bf b0 = lds_load16(Klds + (long)ln * 128 + kk * 32 + 16 * lh);
      v16bf b1 = lds_load16(Klds + (long)(16 + ln) * 128 + kk * 32 + 16 * lh);
      s0 = wmma_bf16(qf[kk], b0, s0);
      s1 = wmma_bf16(qf[kk], b1, s1);
    }

    // --- online softmax (row = r + 8*lh lives in lanes with same lh) ---
#pragma unroll
    for (int r = 0; r < 8; ++r) {
      float mt = fmaxf(s0[r], s1[r]);
#pragma unroll
      for (int off = 1; off < 16; off <<= 1) mt = fmaxf(mt, __shfl_xor(mt, off));
      float mnew  = fmaxf(mrun[r], mt);
      float alpha = __expf((mrun[r] - mnew) * SCALE);
      mrun[r] = mnew;
      float p0 = __expf((s0[r] - mnew) * SCALE);
      float p1 = __expf((s1[r] - mnew) * SCALE);
      float lsum = p0 + p1;
#pragma unroll
      for (int off = 1; off < 16; off <<= 1) lsum += __shfl_xor(lsum, off);
      lrun[r] = lrun[r] * alpha + lsum;
#pragma unroll
      for (int nt = 0; nt < DV / 16; ++nt) o[nt][r] *= alpha;
      pl[(r + 8 * lh) * 32 + ln]      = (bf16)p0;
      pl[(r + 8 * lh) * 32 + 16 + ln] = (bf16)p1;
    }
    __syncthreads();   // P stores visible; waves in lockstep

    // --- O += P @ V ---
    v16bf ap;
#pragma unroll
    for (int j = 0; j < 16; ++j) ap[j] = pl[ln * 32 + kpatA(j, lh)];
#pragma unroll
    for (int nt = 0; nt < DV / 16; ++nt) {
      v16bf bv = lds_load16(Vlds + (long)(nt * 16 + ln) * 32 + 16 * lh);
      o[nt] = wmma_bf16(ap, bv, o[nt]);
    }
    __syncthreads();   // protect K/V LDS before next staging
  }

  // --- normalize and store channel-major ---
  float* ob = O + (long)s * oSS;
#pragma unroll
  for (int r = 0; r < 8; ++r) {
    float inv = 1.f / lrun[r];
    int prow = q0 + r + 8 * lh;
#pragma unroll
    for (int nt = 0; nt < DV / 16; ++nt)
      ob[(long)(nt * 16 + ln) * hw + prow] = o[nt][r] * inv;
  }
}

// ---------------------------------------------------------------------------
// Residual + InstanceNorm2d (no affine) per (s, ch) over 1024 spatial points.
// Everything channel-major; block = 256 threads = one (s, ch) pair.
// ---------------------------------------------------------------------------
__global__ void __launch_bounds__(256)
add_inorm_kernel(const float* __restrict__ tgt, const float* __restrict__ O1,
                 float* __restrict__ Y) {
  const int ch = blockIdx.x, s = blockIdx.y;
  const long base = ((long)s * 256 + ch) * 1024;
  const float* a = tgt + base;
  const float* b = O1 + base;
  float v[4], sum = 0.f, ss = 0.f;
#pragma unroll
  for (int i = 0; i < 4; ++i) {
    int p = threadIdx.x + i * 256;
    v[i] = a[p] + b[p];
    sum += v[i]; ss += v[i] * v[i];
  }
#pragma unroll
  for (int off = 1; off < 32; off <<= 1) {
    sum += __shfl_xor(sum, off);
    ss  += __shfl_xor(ss, off);
  }
  __shared__ float rs[8], rq[8];
  int wv = threadIdx.x >> 5;
  if ((threadIdx.x & 31) == 0) { rs[wv] = sum; rq[wv] = ss; }
  __syncthreads();
  float tsum = 0.f, tss = 0.f;
#pragma unroll
  for (int i = 0; i < 8; ++i) { tsum += rs[i]; tss += rq[i]; }
  float mu   = tsum * (1.f / 1024.f);
  float var  = tss * (1.f / 1024.f) - mu * mu;
  float rsig = rsqrtf(var + 1e-5f);
#pragma unroll
  for (int i = 0; i < 4; ++i) {
    int p = threadIdx.x + i * 256;
    Y[base + p] = (v[i] - mu) * rsig;
  }
}

// ---------------------------------------------------------------------------
extern "C" void kernel_launch(void* const* d_in, const int* in_sizes, int n_in,
                              void* d_out, int out_size, void* d_ws, size_t ws_size,
                              hipStream_t stream) {
  const float* tgt    = (const float*)d_in[0];  // (1,4,256,32,32)
  const float* mem    = (const float*)d_in[1];  // (8,4,256,32,32)
  const float* pos    = (const float*)d_in[2];  // (8,4,128,32,32)
  const float* Wself  = (const float*)d_in[3];  // (128,256)
  const float* bself  = (const float*)d_in[4];  // (128)
  const float* Wcross = (const float*)d_in[5];  // (128,256)
  const float* bcross = (const float*)d_in[6];  // (128)

  const int HW = 1024, C = 256, CE = 128, S = 4, T = 8;

  char* ws = (char*)d_ws;
  bf16*  P1 = (bf16*)(ws);                          // 4*1024*128 bf16 = 1MB
  bf16*  K2 = (bf16*)(ws + (size_t)(1  << 20));     // 4*8192*128 bf16 = 8MB
  float* O1 = (float*)(ws + (size_t)(9  << 20));    // 4*256*1024 f32  = 4MB
  float* Y  = (float*)(ws + (size_t)(13 << 20));    // 4*256*1024 f32  = 4MB
  bf16*  P2 = (bf16*)(ws + (size_t)(17 << 20));     // 4*1024*128 bf16 = 1MB

  // 1) self-attn Q=K projection (cosine normalized)
  proj_l2_kernel<<<dim3(16, S), 128, 0, stream>>>(
      tgt, Wself, bself, P1, HW, 0L, (long)C * HW, (long)HW * 128);
  // 2) cross-attn K projection over memory (8192 rows / batch)
  proj_l2_kernel<<<dim3(128, S), 128, 0, stream>>>(
      mem, Wcross, bcross, K2, HW, (long)S * C * HW, (long)C * HW,
      (long)T * HW * 128);
  // 3) self attention: V = tgt (dv=256), out channel-major fp32
  attn_kernel<256><<<dim3(16, S), 128, 0, stream>>>(
      P1, P1, tgt, O1, HW, (long)HW * 128, (long)HW * 128,
      0L, (long)C * HW, HW, (long)C * HW);
  // 4) residual + instance norm -> Y (channel-major fp32)
  add_inorm_kernel<<<dim3(C, S), 256, 0, stream>>>(tgt, O1, Y);
  // 5) cross-attn Q projection from Y
  proj_l2_kernel<<<dim3(16, S), 128, 0, stream>>>(
      Y, Wcross, bcross, P2, HW, 0L, (long)C * HW, (long)HW * 128);
  // 6) cross attention: V = pos_enc (dv=128), writes final output
  attn_kernel<128><<<dim3(16, S), 128, 0, stream>>>(
      P2, K2, pos, (float*)d_out, T * HW, (long)HW * 128,
      (long)T * HW * 128, (long)S * CE * HW, (long)CE * HW, HW,
      (long)CE * HW);
}
